// GNNUnsupervised_69020124447039
// MI455X (gfx1250) — matrix operand order repaired and checked by reading, not compiled, not run on hardware
//
#include <hip/hip_runtime.h>

#define BATCH  8
#define NNODES 50000
#define NEDGES 400000
#define ROWS   (BATCH * NNODES)   // 400000, divisible by 16*8

typedef long long i64;
typedef __attribute__((ext_vector_type(16))) _Float16 v16h;
typedef __attribute__((ext_vector_type(8)))  _Float16 v8h;
typedef __attribute__((ext_vector_type(4)))  _Float16 v4h;
typedef __attribute__((ext_vector_type(8)))  float    v8f;

// ---------------- elementwise / graph kernels ----------------

__global__ __launch_bounds__(256) void fill_zero4(float4* __restrict__ p, i64 n4) {
  i64 i = (i64)blockIdx.x * blockDim.x + threadIdx.x;
  if (i < n4) p[i] = make_float4(0.0f, 0.0f, 0.0f, 0.0f);
}

// f32 -> f16, 4 elements per thread
__global__ __launch_bounds__(256) void cvt_f32_f16_4(const float4* __restrict__ in,
                                                     v4h* __restrict__ out, i64 n4) {
  i64 i = (i64)blockIdx.x * blockDim.x + threadIdx.x;
  if (i >= n4) return;
  const float4 p = in[i];
  v4h h;
  h[0] = (_Float16)p.x; h[1] = (_Float16)p.y;
  h[2] = (_Float16)p.z; h[3] = (_Float16)p.w;
  out[i] = h;
}

__global__ __launch_bounds__(256) void deg_kernel(const int* __restrict__ cols,
                                                  float* __restrict__ deg) {
  int e = blockIdx.x * blockDim.x + threadIdx.x;
  if (e < NEDGES) atomicAdd(&deg[cols[e]], 1.0f);
}

__global__ __launch_bounds__(256) void norm_kernel(const int* __restrict__ rows,
                                                   const int* __restrict__ cols,
                                                   const float* __restrict__ deg,
                                                   float* __restrict__ nrm) {
  int e = blockIdx.x * blockDim.x + threadIdx.x;
  if (e >= NEDGES) return;
  float dr = deg[rows[e]];
  float dc = deg[cols[e]];
  float a = dr > 0.0f ? rsqrtf(dr) : 0.0f;
  float b = dc > 0.0f ? rsqrtf(dc) : 0.0f;
  nrm[e] = a * b;
}

// h_out[b, col[e], f..f+3] += norm[e] * h_in[b, row[e], f..f+3]   (f16 data,
// scatter via CDNA5 packed-f16 atomics: 2x global_atomic_pk_add_f16 per 4 ch)
template<int F>
__global__ __launch_bounds__(256) void propagate_h(const _Float16* __restrict__ hin,
                                                   _Float16* __restrict__ hout,
                                                   const int* __restrict__ rows,
                                                   const int* __restrict__ cols,
                                                   const float* __restrict__ nrm) {
  constexpr int FV = F / 4;
  i64 idx = (i64)blockIdx.x * blockDim.x + threadIdx.x;
  if (idx >= (i64)NEDGES * FV) return;
  int e = (int)(idx / FV);
  int f = (int)(idx - (i64)e * FV) * 4;
  int r = rows[e], c = cols[e];
  float w = nrm[e];
#pragma unroll
  for (int b = 0; b < BATCH; ++b) {
    const v4h v = *(const v4h*)&hin[((i64)b * NNODES + r) * F + f];
    union { _Float16 h[2]; unsigned u; } p0, p1;
    p0.h[0] = (_Float16)(w * (float)v[0]);
    p0.h[1] = (_Float16)(w * (float)v[1]);
    p1.h[0] = (_Float16)(w * (float)v[2]);
    p1.h[1] = (_Float16)(w * (float)v[3]);
    unsigned long long ad =
        (unsigned long long)&hout[((i64)b * NNODES + c) * F + f];
    asm volatile("global_atomic_pk_add_f16 %0, %1, off"
                 :: "v"(ad), "v"(p0.u) : "memory");
    asm volatile("global_atomic_pk_add_f16 %0, %1, off"
                 :: "v"(ad + 4ull), "v"(p1.u) : "memory");
  }
}

// training-mode BatchNorm1d over batch dim (per (n,c) stats across B=8, computed
// in f32) + leaky relu, in place on f16 activations, 4 channels per thread.
__global__ __launch_bounds__(256) void bn_lrelu_h4(v4h* __restrict__ x,
                                                   const float4* __restrict__ gamma,
                                                   const float4* __restrict__ beta,
                                                   i64 nd4) {
  i64 i = (i64)blockIdx.x * blockDim.x + threadIdx.x;
  if (i >= nd4) return;
  float vals[BATCH][4];
#pragma unroll
  for (int b = 0; b < BATCH; ++b) {
    const v4h p = x[(i64)b * nd4 + i];
#pragma unroll
    for (int comp = 0; comp < 4; ++comp) vals[b][comp] = (float)p[comp];
  }
  float4 g = gamma[i], bt = beta[i];
  const float* gp = (const float*)&g;
  const float* bp = (const float*)&bt;
#pragma unroll
  for (int comp = 0; comp < 4; ++comp) {
    float s = 0.0f;
#pragma unroll
    for (int b = 0; b < BATCH; ++b) s += vals[b][comp];
    float m = s * (1.0f / BATCH);
    float v = 0.0f;
#pragma unroll
    for (int b = 0; b < BATCH; ++b) { float t = vals[b][comp] - m; v += t * t; }
    v *= (1.0f / BATCH);
    float sc = gp[comp] * rsqrtf(v + 1e-5f);
#pragma unroll
    for (int b = 0; b < BATCH; ++b) {
      float t = sc * (vals[b][comp] - m) + bp[comp];
      vals[b][comp] = (t >= 0.0f) ? t : 0.01f * t;
    }
  }
#pragma unroll
  for (int b = 0; b < BATCH; ++b) {
    v4h p;
#pragma unroll
    for (int comp = 0; comp < 4; ++comp) p[comp] = (_Float16)vals[b][comp];
    x[(i64)b * nd4 + i] = p;
  }
}

// ------- fused 4-hop WMMA GEMM: OUT = bias + sum_h A_h[R,Fin] * W[h][Fin,Fout] -------
// A operands are f16 (native WMMA input, direct b128 fragment loads, no converts).
// Weight fragments for all 4 hops live in registers; accumulator (f32) touches
// memory exactly once. Non-final layers store f16 activations; the FINAL variant
// applies  vmin + (vmax-vmin)/(1+exp(u))  in-register and writes f32 output.
// blockDim = 128 (4 waves, wave w -> columns [16w,16w+16)); gridDim.x = R/(16*RT).

template<int Fin, int Fout, int RT, bool FINAL>
__global__ __launch_bounds__(128) void wmma_gemm4(const _Float16* __restrict__ A0,
                                                  const _Float16* __restrict__ A1,
                                                  const _Float16* __restrict__ A2,
                                                  const _Float16* __restrict__ A3,
                                                  const float* __restrict__ W,
                                                  const float* __restrict__ bias,
                                                  void* __restrict__ OUTv, int R,
                                                  const float* __restrict__ vmin,
                                                  const float* __restrict__ vmax) {
  constexpr int KCH = (Fin + 31) / 32;   // 32-wide K chunks (1 or 2 here)
  const int lane = threadIdx.x & 31;
  const int wave = threadIdx.x >> 5;
  const int colBase = wave * 16;
  if (colBase >= Fout) return;           // wave-uniform exit (EXEC stays full)
  const int lo16 = lane & 15;
  const bool hi = (lane >= 16);
  // A layout (16-bit A 16x32): element i holds K = aHalf0+i (i<8) / aHalf1+(i-8), i.e. K = 16*hi + i; M = lane&15
  const int aHalf0 = hi ? 16 : 0;
  const int aHalf1 = hi ? 24 : 8;
  // B layout (32x16): element i holds K = bkBase+i, column N = lane&15
  const int bn = colBase + lo16;
  const int bkBase = hi ? 16 : 0;
  const bool colOk = (bn < Fout);
  // C/D layout: VGPR r -> M = cm + r, N = colBase + (lane&15)
  const int cm = hi ? 8 : 0;

  // ---- weight fragments for all 4 hops, loaded once (f32 weights -> f16) ----
  v16h bfrag[4][KCH];
#pragma unroll
  for (int h = 0; h < 4; ++h)
#pragma unroll
    for (int kc = 0; kc < KCH; ++kc)
#pragma unroll
      for (int i = 0; i < 16; ++i) {
        const int k = kc * 32 + bkBase + i;
        if (k < Fin)
          bfrag[h][kc][i] = colOk ? (_Float16)W[h * Fin * Fout + k * Fout + bn]
                                  : (_Float16)0.0f;
        else
          bfrag[h][kc][i] = (_Float16)0.0f;
      }

  const float cinit = colOk ? bias[bn] : 0.0f;
  const _Float16* const As[4] = {A0, A1, A2, A3};
  constexpr _Float16 hzero = (_Float16)0.0f;

  const int tile0 = blockIdx.x * RT;
  for (int t = 0; t < RT; ++t) {
    const int rowBase = (tile0 + t) * 16;
    if (rowBase >= R) return;
    const i64 arow = (i64)(rowBase + lo16) * Fin;

    v8f c;
#pragma unroll
    for (int r = 0; r < 8; ++r) c[r] = cinit;

#pragma unroll
    for (int h = 0; h < 4; ++h) {
      const _Float16* __restrict__ A = As[h];
#pragma unroll
      for (int kc = 0; kc < KCH; ++kc) {
        v16h a;
        if constexpr (Fin % 32 == 0) {
          // dense: two b128 loads of 8 contiguous f16 K-values each, no converts
          const v8h p0 = *(const v8h*)&A[arow + kc * 32 + aHalf0];
          const v8h p1 = *(const v8h*)&A[arow + kc * 32 + aHalf1];
#pragma unroll
          for (int i = 0; i < 8; ++i) { a[i] = p0[i]; a[8 + i] = p1[i]; }
        } else {
          // Fin == 4: one b64 load, valid for every lane; hi lanes zero by select
          const v4h p = *(const v4h*)&A[arow];
#pragma unroll
          for (int i = 4; i < 16; ++i) a[i] = hzero;
          a[0] = hi ? hzero : p[0];
          a[1] = hi ? hzero : p[1];
          a[2] = hi ? hzero : p[2];
          a[3] = hi ? hzero : p[3];
        }
        // (neg_a, A, neg_b, B, c_mod, C, reuse_a, reuse_b)
        c = __builtin_amdgcn_wmma_f32_16x16x32_f16(false, a, false, bfrag[h][kc],
                                                   (short)0, c, false, false);
      }
    }

#pragma unroll
    for (int r = 0; r < 8; ++r) {
      if (colOk) {
        const int row = rowBase + cm + r;          // row = b*NNODES + n
        if constexpr (FINAL) {
          const int n = row % NNODES;
          const int nc = n * Fout + bn;
          const float mn = vmin[nc], mx = vmax[nc];
          ((float*)OUTv)[(i64)row * Fout + bn] = mn + (mx - mn) / (1.0f + expf(c[r]));
        } else {
          ((_Float16*)OUTv)[(i64)row * Fout + bn] = (_Float16)c[r];
        }
      }
    }
  }
}

// ---------------- host orchestration ----------------

static inline int nblk(i64 n, int t) { return (int)((n + t - 1) / t); }

#define RT_TILES 8
#define GEMM_GRID (ROWS / (16 * RT_TILES))   // 3125

extern "C" void kernel_launch(void* const* d_in, const int* in_sizes, int n_in,
                              void* d_out, int out_size, void* d_ws, size_t ws_size,
                              hipStream_t stream) {
  (void)in_sizes; (void)n_in; (void)out_size; (void)ws_size;
  const float* x    = (const float*)d_in[0];
  const int*   ei   = (const int*)d_in[1];     // [2, E]
  const float* vmin = (const float*)d_in[2];
  const float* vmax = (const float*)d_in[3];
  const float* W0 = (const float*)d_in[4];   const float* b0 = (const float*)d_in[5];
  const float* W1 = (const float*)d_in[6];   const float* b1 = (const float*)d_in[7];
  const float* W2 = (const float*)d_in[8];   const float* b2 = (const float*)d_in[9];
  const float* g0 = (const float*)d_in[10];  const float* be0 = (const float*)d_in[11];
  const float* g1 = (const float*)d_in[12];  const float* be1 = (const float*)d_in[13];
  const int* rows = ei;
  const int* cols = ei + NEDGES;

  char* wp = (char*)d_ws;
  auto take = [&](size_t bytes) {
    char* p = wp;
    wp += (bytes + 255) & ~(size_t)255;
    return p;
  };
  const size_t BIGH = (size_t)ROWS * 64 * sizeof(_Float16);   // 51.2 MB
  _Float16* xh   = (_Float16*)take((size_t)ROWS * 4 * sizeof(_Float16));
  _Float16* big0 = (_Float16*)take(BIGH);            // layer0 out / layer1 in
  _Float16* big1 = (_Float16*)take(BIGH);            // hop h1
  _Float16* big2 = (_Float16*)take(BIGH);            // hop h2
  _Float16* big3 = (_Float16*)take(BIGH);            // hop h3
  _Float16* big4 = (_Float16*)take(BIGH);            // layer1 out / layer2 in
  float* deg  = (float*)take((size_t)NNODES * 4);
  float* nrm  = (float*)take((size_t)NEDGES * 4);

  // gcn_norm
  fill_zero4<<<nblk(NNODES / 4, 256), 256, 0, stream>>>((float4*)deg, NNODES / 4);
  deg_kernel<<<nblk(NEDGES, 256), 256, 0, stream>>>(cols, deg);
  norm_kernel<<<nblk(NEDGES, 256), 256, 0, stream>>>(rows, cols, deg, nrm);

  // input features -> f16 once
  cvt_f32_f16_4<<<nblk(ROWS, 256), 256, 0, stream>>>((const float4*)x, (v4h*)xh, ROWS);

  // ---- Layer 0: Fin=4, Fout=64 ----
  {
    constexpr int Fin = 4;
    const i64 z4 = (i64)ROWS * Fin * 2 / 16;           // f16 bytes / 16
    const i64 pThreads = (i64)NEDGES * (Fin / 4);
    fill_zero4<<<nblk(z4, 256), 256, 0, stream>>>((float4*)big1, z4);
    propagate_h<Fin><<<nblk(pThreads, 256), 256, 0, stream>>>(xh, big1, rows, cols, nrm);
    fill_zero4<<<nblk(z4, 256), 256, 0, stream>>>((float4*)big2, z4);
    propagate_h<Fin><<<nblk(pThreads, 256), 256, 0, stream>>>(big1, big2, rows, cols, nrm);
    fill_zero4<<<nblk(z4, 256), 256, 0, stream>>>((float4*)big3, z4);
    propagate_h<Fin><<<nblk(pThreads, 256), 256, 0, stream>>>(big2, big3, rows, cols, nrm);
    wmma_gemm4<Fin, 64, RT_TILES, false><<<GEMM_GRID, 128, 0, stream>>>(
        xh, big1, big2, big3, W0, b0, big0, ROWS, nullptr, nullptr);
    bn_lrelu_h4<<<nblk((i64)NNODES * 16, 256), 256, 0, stream>>>(
        (v4h*)big0, (const float4*)g0, (const float4*)be0, (i64)NNODES * 16);
  }

  // ---- Layer 1: Fin=64, Fout=64 ----
  {
    constexpr int Fin = 64;
    const i64 z4 = (i64)ROWS * Fin * 2 / 16;
    const i64 pThreads = (i64)NEDGES * (Fin / 4);
    fill_zero4<<<nblk(z4, 256), 256, 0, stream>>>((float4*)big1, z4);
    propagate_h<Fin><<<nblk(pThreads, 256), 256, 0, stream>>>(big0, big1, rows, cols, nrm);
    fill_zero4<<<nblk(z4, 256), 256, 0, stream>>>((float4*)big2, z4);
    propagate_h<Fin><<<nblk(pThreads, 256), 256, 0, stream>>>(big1, big2, rows, cols, nrm);
    fill_zero4<<<nblk(z4, 256), 256, 0, stream>>>((float4*)big3, z4);
    propagate_h<Fin><<<nblk(pThreads, 256), 256, 0, stream>>>(big2, big3, rows, cols, nrm);
    wmma_gemm4<Fin, 64, RT_TILES, false><<<GEMM_GRID, 128, 0, stream>>>(
        big0, big1, big2, big3, W1, b1, big4, ROWS, nullptr, nullptr);
    bn_lrelu_h4<<<nblk((i64)NNODES * 16, 256), 256, 0, stream>>>(
        (v4h*)big4, (const float4*)g1, (const float4*)be1, (i64)NNODES * 16);
  }

  // ---- Layer 2: Fin=64, Fout=3, fused range-sigmoid epilogue -> f32 d_out ----
  {
    constexpr int Fin = 64;
    const i64 z4 = (i64)ROWS * Fin * 2 / 16;
    const i64 pThreads = (i64)NEDGES * (Fin / 4);
    fill_zero4<<<nblk(z4, 256), 256, 0, stream>>>((float4*)big1, z4);
    propagate_h<Fin><<<nblk(pThreads, 256), 256, 0, stream>>>(big4, big1, rows, cols, nrm);
    fill_zero4<<<nblk(z4, 256), 256, 0, stream>>>((float4*)big2, z4);
    propagate_h<Fin><<<nblk(pThreads, 256), 256, 0, stream>>>(big1, big2, rows, cols, nrm);
    fill_zero4<<<nblk(z4, 256), 256, 0, stream>>>((float4*)big3, z4);
    propagate_h<Fin><<<nblk(pThreads, 256), 256, 0, stream>>>(big2, big3, rows, cols, nrm);
    wmma_gemm4<Fin, 3, RT_TILES, true><<<GEMM_GRID, 128, 0, stream>>>(
        big4, big1, big2, big3, W2, b2, d_out, ROWS, vmin, vmax);
  }
}
